// SurrogateModel_6236292514024
// MI455X (gfx1250) — compile-verified, matrix operand order checked by reading, not run on hardware
//
#include <hip/hip_runtime.h>
#include <math.h>

typedef __attribute__((ext_vector_type(16))) _Float16 v16h;
typedef __attribute__((ext_vector_type(8)))  float    v8f;

// ---------------------------------------------------------------------------
// Utility kernels
// ---------------------------------------------------------------------------
__global__ void k_fill(float* __restrict__ p, float v, int n) {
    int i = blockIdx.x * blockDim.x + threadIdx.x;
    if (i < n) p[i] = v;
}

__global__ void k_deg(const long long* __restrict__ dst, float* __restrict__ deg, int E) {
    int e = blockIdx.x * blockDim.x + threadIdx.x;
    if (e < E) atomicAdd(&deg[(int)dst[e]], 1.0f);
}

__global__ void k_rsqrt_inplace(float* __restrict__ d, int n) {
    int i = blockIdx.x * blockDim.x + threadIdx.x;
    if (i < n) d[i] = rsqrtf(d[i]);   // deg >= 1 always (self loops)
}

__global__ void k_cvt_f16(const float* __restrict__ in, _Float16* __restrict__ out, int n) {
    int i = blockIdx.x * blockDim.x + threadIdx.x;
    if (i < n) out[i] = (_Float16)in[i];
}

// W row-major [K][Nw] (f32)  ->  Wt [Npad][K] (f16), zero-padded cols
__global__ void k_transpose_cvt(const float* __restrict__ W, _Float16* __restrict__ Wt,
                                int K, int Nw, int Npad) {
    int idx = blockIdx.x * blockDim.x + threadIdx.x;
    if (idx >= K * Npad) return;
    int n = idx / K;
    int k = idx - n * K;
    Wt[(size_t)n * K + k] = (n < Nw) ? (_Float16)W[(size_t)k * Nw + n] : (_Float16)0.0f;
}

// ---------------------------------------------------------------------------
// WMMA GEMM, K fixed at 128: C[M x Nout] (f32) = A[M x 128] (f16) * B
//   Bt stored as [Npad][128] f16 (B transposed) -> contiguous K pairs.
//   One wave32 per 16-row strip: the full K-extent of A (4 fragments,
//   32 VGPRs) is held in registers and reused across ALL NT column tiles
//   (NT*4 WMMAs per wave, A fetched exactly once).
//   Block = 128 threads = 4 waves = 4 consecutive M tiles.
// ---------------------------------------------------------------------------
template <int NT>
__global__ void k_wmma_gemm(const _Float16* __restrict__ A,
                            const _Float16* __restrict__ Bt,
                            float* __restrict__ C,
                            int M, int Nout) {
    const int lane = threadIdx.x & 31;                     // wave32 lane
    const int tm = blockIdx.x * 4 + (threadIdx.x >> 5);    // M tile per wave
    if ((tm + 1) * 16 > M) return;                         // wave-uniform guard

    const int mi = lane & 15;         // row (A) / col (B) within tile
    const int kh = (lane >> 4) & 1;   // K-half select (ISA 16-bit frag layout)

    const _Float16* __restrict__ arow = A + (size_t)(tm * 16 + mi) * 128;

    // Load entire K=128 extent of A once: 4 fragments of 16 f16 per lane.
    v16h a[4];
#pragma unroll
    for (int s = 0; s < 4; ++s) {
#pragma unroll
        for (int v = 0; v < 8; ++v) {
            // VGPR v holds K = (v/4)*16 + kh*8 + (v%4)*2 (+0,+1), plus 32*s
            int k0 = s * 32 + ((v >> 2) << 4) + (kh << 3) + ((v & 3) << 1);
            a[s][2 * v]     = arow[k0];
            a[s][2 * v + 1] = arow[k0 + 1];
        }
    }

#pragma unroll
    for (int tn = 0; tn < NT; ++tn) {
        const _Float16* __restrict__ brow = Bt + (size_t)(tn * 16 + mi) * 128;
        v8f acc = {};
#pragma unroll
        for (int s = 0; s < 4; ++s) {
            v16h b;
#pragma unroll
            for (int v = 0; v < 8; ++v) {
                int k0 = s * 32 + ((v >> 2) << 4) + (kh << 3) + ((v & 3) << 1);
                b[2 * v]     = brow[k0];
                b[2 * v + 1] = brow[k0 + 1];
            }
            acc = __builtin_amdgcn_wmma_f32_16x16x32_f16(
                /*neg_a=*/false, a[s], /*neg_b=*/false, b,
                /*c_mod=*/(short)0, acc, /*reuse_a=*/false, /*reuse_b=*/false);
        }
#pragma unroll
        for (int v = 0; v < 8; ++v) {
            // C/D layout: VGPR v -> M = v (lanes 0-15) or 8+v (lanes 16-31)
            int m = tm * 16 + v + (kh << 3);
            int n = tn * 16 + mi;
            if (n < Nout) C[(size_t)m * Nout + n] = acc[v];
        }
    }
}

// ---------------------------------------------------------------------------
// Edge scatter, F=128: one wave per edge, float4 per lane, fp32 atomics (L2).
// ---------------------------------------------------------------------------
__global__ void k_scatter128(const long long* __restrict__ src,
                             const long long* __restrict__ dst,
                             const float* __restrict__ dinv,
                             const float* __restrict__ h,
                             float* __restrict__ agg, int E) {
    int t = blockIdx.x * blockDim.x + threadIdx.x;
    int e = t >> 5;
    int lane = t & 31;
    if (e >= E) return;
    int s = (int)src[e];
    int d = (int)dst[e];
    float nm = dinv[s] * dinv[d];
    const float4 hv = reinterpret_cast<const float4*>(h + (size_t)s * 128)[lane];
    float* ap = agg + (size_t)d * 128 + lane * 4;
    atomicAdd(ap + 0, hv.x * nm);
    atomicAdd(ap + 1, hv.y * nm);
    atomicAdd(ap + 2, hv.z * nm);
    atomicAdd(ap + 3, hv.w * nm);
}

// agg + self-loop (norm = dinv^2) + bias, ReLU, convert to f16 for next GEMM
__global__ void k_bias_relu_cvt(const float* __restrict__ agg,
                                const float* __restrict__ hlin,
                                const float* __restrict__ dinv,
                                const float* __restrict__ b,
                                _Float16* __restrict__ out, int total) {
    int idx = blockIdx.x * blockDim.x + threadIdx.x;
    if (idx >= total) return;
    int i = idx >> 7;
    int f = idx & 127;
    float di = dinv[i];
    float v = agg[idx] + hlin[idx] * di * di + b[f];
    out[idx] = (_Float16)fmaxf(v, 0.0f);
}

// Edge scatter, F=40: one thread per (edge, feature)
__global__ void k_scatter40(const long long* __restrict__ src,
                            const long long* __restrict__ dst,
                            const float* __restrict__ dinv,
                            const float* __restrict__ h,
                            float* __restrict__ agg, int total) {
    int idx = blockIdx.x * blockDim.x + threadIdx.x;
    if (idx >= total) return;
    int e = idx / 40;
    int f = idx - e * 40;
    int s = (int)src[e];
    int d = (int)dst[e];
    float nm = dinv[s] * dinv[d];
    atomicAdd(agg + (size_t)d * 40 + f, h[(size_t)s * 40 + f] * nm);
}

// Final: val = agg2 + self-loop + b2, then row-wise log_softmax over 40 classes.
// One wave32 per node: lane covers class f=lane and f=lane+32 (lane<8).
__global__ void k_logsoftmax(const float* __restrict__ agg,
                             const float* __restrict__ hlin,
                             const float* __restrict__ dinv,
                             const float* __restrict__ b2,
                             float* __restrict__ out, int N) {
    int node = blockIdx.x * (blockDim.x >> 5) + (threadIdx.x >> 5);
    int lane = threadIdx.x & 31;
    if (node >= N) return;
    float di = dinv[node];
    float nn = di * di;
    const float* a = agg  + (size_t)node * 40;
    const float* h = hlin + (size_t)node * 40;
    float v0 = a[lane] + h[lane] * nn + b2[lane];
    float v1 = (lane < 8) ? (a[32 + lane] + h[32 + lane] * nn + b2[32 + lane])
                          : -3.402823466e38f;
    float m = fmaxf(v0, v1);
    for (int off = 16; off >= 1; off >>= 1) m = fmaxf(m, __shfl_xor(m, off, 32));
    float s = expf(v0 - m) + ((lane < 8) ? expf(v1 - m) : 0.0f);
    for (int off = 16; off >= 1; off >>= 1) s += __shfl_xor(s, off, 32);
    float ls = m + logf(s);
    out[(size_t)node * 40 + lane] = v0 - ls;
    if (lane < 8) out[(size_t)node * 40 + 32 + lane] = v1 - ls;
}

// ---------------------------------------------------------------------------
// Orchestration
// ---------------------------------------------------------------------------
extern "C" void kernel_launch(void* const* d_in, const int* in_sizes, int n_in,
                              void* d_out, int out_size, void* d_ws, size_t ws_size,
                              hipStream_t stream) {
    const float*     x   = (const float*)d_in[0];
    const long long* ei  = (const long long*)d_in[1];
    const float*     W1  = (const float*)d_in[2];
    const float*     b1  = (const float*)d_in[3];
    const float*     W2  = (const float*)d_in[4];
    const float*     b2  = (const float*)d_in[5];
    float*           out = (float*)d_out;

    const int N = in_sizes[0] / 128;     // 50000
    const int E = in_sizes[1] / 2;       // 800000
    const long long* esrc = ei;
    const long long* edst = ei + E;

    // Carve workspace (256B-aligned regions)
    char* w = (char*)d_ws;
    size_t off = 0;
    auto carve = [&](size_t bytes) -> void* {
        void* p = w + off;
        off += (bytes + 255) & ~(size_t)255;
        return p;
    };
    float*    dinv = (float*)   carve((size_t)N * 4);            // deg -> dinv (in place)
    _Float16* xh   = (_Float16*)carve((size_t)N * 128 * 2);      // x f16; reused as h1h
    _Float16* w1t  = (_Float16*)carve((size_t)128 * 128 * 2);    // W1^T f16
    _Float16* w2t  = (_Float16*)carve((size_t)48 * 128 * 2);     // W2^T f16 (padded to 48)
    float*    hlin = (float*)   carve((size_t)N * 128 * 4);      // layer1 pre-agg; reused as h2lin
    float*    agg  = (float*)   carve((size_t)N * 128 * 4);      // agg1; reused as agg2
    _Float16* h1h   = xh;      // reuse (xh dead after GEMM1)
    float*    h2lin = hlin;    // reuse (hlin dead after bias_relu)
    float*    agg2  = agg;     // reuse (agg1 dead after bias_relu)

    const int BT = 256;
    const int mtiles = N / 16;                 // 3125 (N is a multiple of 16)
    const int gemmBlocks = (mtiles + 3) / 4;   // 4 waves (M tiles) per block

    // ---- normalization: deg (self-loop=1) -> atomic count -> rsqrt ----
    k_fill<<<(N + BT - 1) / BT, BT, 0, stream>>>(dinv, 1.0f, N);
    k_deg<<<(E + BT - 1) / BT, BT, 0, stream>>>(edst, dinv, E);
    k_rsqrt_inplace<<<(N + BT - 1) / BT, BT, 0, stream>>>(dinv, N);

    // ---- convert inputs to f16, transpose weights ----
    k_cvt_f16<<<((N * 128) + BT - 1) / BT, BT, 0, stream>>>(x, xh, N * 128);
    k_transpose_cvt<<<(128 * 128 + BT - 1) / BT, BT, 0, stream>>>(W1, w1t, 128, 128, 128);
    k_transpose_cvt<<<(128 * 48 + BT - 1) / BT, BT, 0, stream>>>(W2, w2t, 128, 40, 48);

    // ---- layer 1: GEMM (WMMA) -> edge scatter -> bias+selfloop+ReLU ----
    k_fill<<<((N * 128) + BT - 1) / BT, BT, 0, stream>>>(agg, 0.0f, N * 128);
    k_wmma_gemm<8><<<gemmBlocks, 128, 0, stream>>>(xh, w1t, hlin, N, 128);
    k_scatter128<<<((E * 32) + BT - 1) / BT, BT, 0, stream>>>(esrc, edst, dinv, hlin, agg, E);
    k_bias_relu_cvt<<<((N * 128) + BT - 1) / BT, BT, 0, stream>>>(agg, hlin, dinv, b1, h1h, N * 128);

    // ---- layer 2: GEMM (WMMA, N padded 40->48) -> scatter -> log_softmax ----
    k_wmma_gemm<3><<<gemmBlocks, 128, 0, stream>>>(h1h, w2t, h2lin, N, 40);
    k_fill<<<((N * 40) + BT - 1) / BT, BT, 0, stream>>>(agg2, 0.0f, N * 40);
    k_scatter40<<<((E * 40) + BT - 1) / BT, BT, 0, stream>>>(esrc, edst, dinv, h2lin, agg2, E * 40);
    k_logsoftmax<<<((N + 7) / 8), 256, 0, stream>>>(agg2, h2lin, dinv, b2, out, N);
}
